// _MockGQAAttention_71519795413047
// MI455X (gfx1250) — matrix operand order repaired and checked by reading, not compiled, not run on hardware
//
#include <hip/hip_runtime.h>
#include <hip/hip_bf16.h>

// ---------------------------------------------------------------------------
// MI455X (gfx1250) GQA attention: bf16 WMMA pipeline, f32 in/out.
//   S=2048, HIDDEN=4096, HQ=32, HKV=8, D=128, n_rep=4
// ---------------------------------------------------------------------------

#define S_LEN   2048
#define HID     4096
#define NQH     32
#define NKVH    8
#define HD      128

typedef __bf16 bf16_t;
typedef bf16_t v8bf   __attribute__((ext_vector_type(8)));
typedef bf16_t v16bf  __attribute__((ext_vector_type(16)));
typedef float  v8f    __attribute__((ext_vector_type(8)));

static __device__ __forceinline__ v8bf load8(const bf16_t* p) {
    return *reinterpret_cast<const v8bf*>(p);
}
static __device__ __forceinline__ v16bf cat16(v8bf lo, v8bf hi) {
    return __builtin_shufflevector(lo, hi, 0,1,2,3,4,5,6,7,8,9,10,11,12,13,14,15);
}
static __device__ __forceinline__ v8f wmma_bf16(v16bf a, v16bf b, v8f c) {
    return __builtin_amdgcn_wmma_f32_16x16x32_bf16(
        /*neg_a=*/false, a, /*neg_b=*/false, b,
        /*c_mod=*/(short)0, c, /*reuse_a=*/false, /*reuse_b=*/false);
}

// A-fragment (16x32 bf16, M x K) from row-major [*, lda], tile origin (r0, k0).
// Lane L (half = L>>4, r = L&15): elems j -> k = k0 + j + 16*half
static __device__ __forceinline__ v16bf load_afrag(const bf16_t* base, size_t lda,
                                                   int r0, int k0, int lane) {
    const int half = lane >> 4, r = lane & 15;
    const bf16_t* p = base + (size_t)(r0 + r) * lda + k0;
    return cat16(load8(p + 16 * half), load8(p + 16 * half + 8));
}

// B-fragment (32x16 bf16, K x N) with B stored column-major, i.e. as
// BT row-major [N, ldk]: lane n -> col n&15, elems j -> k = k0 + j + 16*(n>>4).
static __device__ __forceinline__ v16bf load_bfrag_colmajor(const bf16_t* bt, size_t ldk,
                                                            int n0, int k0, int lane) {
    const int half = lane >> 4, c = lane & 15;
    const bf16_t* p = bt + (size_t)(n0 + c) * ldk + k0 + 16 * half;
    return cat16(load8(p), load8(p + 8));
}

// ---------------------------------------------------------------------------
// 1) elementwise f32 -> bf16
// ---------------------------------------------------------------------------
__global__ void k_cvt_f32_bf16(const float* __restrict__ in, bf16_t* __restrict__ out, int n) {
    int i = blockIdx.x * blockDim.x + threadIdx.x;
    int stride = gridDim.x * blockDim.x;
    for (; i < n; i += stride) out[i] = (bf16_t)in[i];
}

// ---------------------------------------------------------------------------
// 2) tiled transpose + convert:  in f32 [R, C] row-major -> out bf16 [C, R]
// ---------------------------------------------------------------------------
__global__ void k_transpose_f32_bf16(const float* __restrict__ in, bf16_t* __restrict__ out,
                                     int R, int C) {
    __shared__ float tile[32][33];
    const int bx = blockIdx.x, by = blockIdx.y;
    const int x = bx * 32 + threadIdx.x;
    #pragma unroll
    for (int i = threadIdx.y; i < 32; i += 8) {
        int y = by * 32 + i;
        tile[i][threadIdx.x] = (y < R && x < C) ? in[(size_t)y * C + x] : 0.f;
    }
    __syncthreads();
    const int xo = by * 32 + threadIdx.x;          // out col == in row
    #pragma unroll
    for (int i = threadIdx.y; i < 32; i += 8) {
        int yo = bx * 32 + i;                      // out row == in col
        if (yo < C && xo < R) out[(size_t)yo * R + xo] = (bf16_t)tile[threadIdx.x][i];
    }
}

// ---------------------------------------------------------------------------
// 3/5) bf16 GEMM: C[M,N] = A[M,K] * B[K,N], B given as BT[N,K] (col-major B).
// Block 256 thr = 8 waves in 2(M) x 4(N); wave tile 64x64 (4x4 fragments,
// 16 WMMAs per 32-k step); block tile 128x256.
// MODE 0: f32 row-major out (ldc)   MODE 1: bf16 row-major out (ldc)
// MODE 2: bf16 out transposed per-KV-head: VT[kv][128][S_LEN]
// ---------------------------------------------------------------------------
template <int MODE>
__global__ __launch_bounds__(256)
void k_gemm_bf16(const bf16_t* __restrict__ A, const bf16_t* __restrict__ BT,
                 void* __restrict__ outp, int M, int N, int K, int ldc) {
    const int lane = threadIdx.x & 31;
    const int wave = threadIdx.x >> 5;
    const int wy = wave >> 2, wx = wave & 3;
    const int half = lane >> 4, hl = lane & 15;
    const int m0 = blockIdx.y * 128 + wy * 64;
    const int n0 = blockIdx.x * 256 + wx * 64;
    (void)M; (void)N;

    v8f acc[4][4];
    #pragma unroll
    for (int ii = 0; ii < 4; ++ii)
        #pragma unroll
        for (int j = 0; j < 4; ++j)
            acc[ii][j] = v8f{0.f,0.f,0.f,0.f,0.f,0.f,0.f,0.f};

    for (int k0 = 0; k0 < K; k0 += 32) {
        v16bf a[4], b[4];
        #pragma unroll
        for (int ii = 0; ii < 4; ++ii)
            a[ii] = load_afrag(A, (size_t)K, m0 + 16 * ii, k0, lane);
        #pragma unroll
        for (int j = 0; j < 4; ++j)
            b[j] = load_bfrag_colmajor(BT, (size_t)K, n0 + 16 * j, k0, lane);
        #pragma unroll
        for (int ii = 0; ii < 4; ++ii)
            #pragma unroll
            for (int j = 0; j < 4; ++j)
                acc[ii][j] = wmma_bf16(a[ii], b[j], acc[ii][j]);
    }

    #pragma unroll
    for (int ii = 0; ii < 4; ++ii) {
        #pragma unroll
        for (int j = 0; j < 4; ++j) {
            const int col = n0 + 16 * j + hl;
            #pragma unroll
            for (int i = 0; i < 8; ++i) {
                const int row = m0 + 16 * ii + i + 8 * half;
                if (MODE == 0) {
                    ((float*)outp)[(size_t)row * ldc + col] = acc[ii][j][i];
                } else if (MODE == 1) {
                    ((bf16_t*)outp)[(size_t)row * ldc + col] = (bf16_t)acc[ii][j][i];
                } else {
                    const int kv = col >> 7, d = col & 127;
                    ((bf16_t*)outp)[(size_t)kv * (HD * S_LEN) + (size_t)d * S_LEN + row] =
                        (bf16_t)acc[ii][j][i];
                }
            }
        }
    }
}

// ---------------------------------------------------------------------------
// 4) flash attention. One wave per (head, 16-query tile). 8 waves / block.
//    Q  : bf16 [S, NQH*HD] row-major
//    K  : bf16 [S, NKVH*HD] row-major (acts as col-major B for Q@K^T)
//    VT : bf16 [NKVH][HD][S] (acts as col-major B for P@V)
//    O  : bf16 [S, NQH*HD]
// Row softmax stats: max reduced across the 16-lane half each block;
// the row-sum l is carried as a per-lane partial and reduced once at the end
// (corr is uniform across the half, so partial sums rescale uniformly).
// ---------------------------------------------------------------------------
__global__ __launch_bounds__(256)
void k_flash_attn(const bf16_t* __restrict__ Qb, const bf16_t* __restrict__ Kb,
                  const bf16_t* __restrict__ VTb, bf16_t* __restrict__ Ob) {
    __shared__ __attribute__((aligned(16))) bf16_t plds[8 * 16 * 32];
    const int lane = threadIdx.x & 31;
    const int wave = threadIdx.x >> 5;
    const int gw = blockIdx.x * 8 + wave;        // 0 .. 4095
    const int h  = gw >> 7;                      // head (128 q-tiles per head)
    const int qt = gw & 127;
    const int q0 = qt * 16;
    const int kvh = h >> 2;                      // n_rep = 4
    const int half = lane >> 4, hl = lane & 15;
    bf16_t* P = plds + wave * (16 * 32);

    // Q A-fragments: 16 queries x 128 dims = 4 fragments of K=32
    v16bf qa[4];
    #pragma unroll
    for (int df = 0; df < 4; ++df)
        qa[df] = load_afrag(Qb, (size_t)(NQH * HD), q0, h * HD + df * 32, lane);

    v8f o[8];
    #pragma unroll
    for (int f = 0; f < 8; ++f) o[f] = v8f{0.f,0.f,0.f,0.f,0.f,0.f,0.f,0.f};
    float m_i[8], l_i[8];                        // l_i: per-lane partial row sum
    #pragma unroll
    for (int i = 0; i < 8; ++i) { m_i[i] = -1e30f; l_i[i] = 0.f; }

    const float scale = 0.08838834764831845f;    // 1/sqrt(128)

    for (int kb = 0; kb < q0 + 16; kb += 32) {
        // ---- scores: 16 q x 32 keys ----
        v8f s0 = v8f{0.f,0.f,0.f,0.f,0.f,0.f,0.f,0.f};
        v8f s1 = v8f{0.f,0.f,0.f,0.f,0.f,0.f,0.f,0.f};
        #pragma unroll
        for (int df = 0; df < 4; ++df) {
            // B[d][key]: K rows are contiguous in d => col-major B directly
            const int dcol = kvh * HD + df * 32;
            const bf16_t* kp0 = Kb + (size_t)(kb + hl) * (NKVH * HD) + dcol + 16 * half;
            const bf16_t* kp1 = kp0 + (size_t)16 * (NKVH * HD);
            v16bf b0 = cat16(load8(kp0), load8(kp0 + 8));
            v16bf b1 = cat16(load8(kp1), load8(kp1 + 8));
            s0 = wmma_bf16(qa[df], b0, s0);
            s1 = wmma_bf16(qa[df], b1, s1);
        }
        // ---- online softmax update ----
        #pragma unroll
        for (int i = 0; i < 8; ++i) {
            const int qrow = q0 + i + 8 * half;
            const int c0 = kb + hl, c1 = c0 + 16;
            float x0 = (c0 <= qrow) ? s0[i] * scale : -1e30f;
            float x1 = (c1 <= qrow) ? s1[i] * scale : -1e30f;
            float rmax = fmaxf(x0, x1);
            #pragma unroll
            for (int off = 1; off < 16; off <<= 1)
                rmax = fmaxf(rmax, __shfl_xor(rmax, off, 32));
            const float mn = fmaxf(m_i[i], rmax);
            const float corr = __expf(m_i[i] - mn);
            const float p0 = __expf(x0 - mn);
            const float p1 = __expf(x1 - mn);
            l_i[i] = l_i[i] * corr + (p0 + p1);  // per-lane partial; reduced later
            m_i[i] = mn;
            #pragma unroll
            for (int f = 0; f < 8; ++f) o[f][i] *= corr;
            // stage P (16x32 bf16) to this wave's private LDS slice
            P[(i + 8 * half) * 32 + hl]      = (bf16_t)p0;
            P[(i + 8 * half) * 32 + hl + 16] = (bf16_t)p1;
        }
        // per-wave LDS ops are in-order; wait + compiler fence before re-reading
        asm volatile("s_wait_dscnt 0" ::: "memory");
        const bf16_t* pr = P + hl * 32;
        v16bf pa = cat16(load8(pr + 16 * half), load8(pr + 16 * half + 8));
        asm volatile("" ::: "memory");
        // ---- P @ V : accumulate 16 q x 128 d ----
        #pragma unroll
        for (int f = 0; f < 8; ++f) {
            const bf16_t* vp = VTb + (size_t)kvh * (HD * S_LEN)
                                   + (size_t)(f * 16 + hl) * S_LEN + kb + 16 * half;
            v16bf bv = cat16(load8(vp), load8(vp + 8));
            o[f] = wmma_bf16(pa, bv, o[f]);
        }
        asm volatile("" ::: "memory");
    }

    // final row-sum reduction across the 16-lane half, then normalize
    #pragma unroll
    for (int i = 0; i < 8; ++i) {
        float rsum = l_i[i];
        #pragma unroll
        for (int off = 1; off < 16; off <<= 1)
            rsum += __shfl_xor(rsum, off, 32);
        l_i[i] = 1.f / rsum;
    }
    #pragma unroll
    for (int f = 0; f < 8; ++f) {
        #pragma unroll
        for (int i = 0; i < 8; ++i) {
            const int row = q0 + i + 8 * half;
            Ob[(size_t)row * (NQH * HD) + h * HD + f * 16 + hl] =
                (bf16_t)(o[f][i] * l_i[i]);
        }
    }
}

// ---------------------------------------------------------------------------
// host orchestration
// ---------------------------------------------------------------------------
extern "C" void kernel_launch(void* const* d_in, const int* in_sizes, int n_in,
                              void* d_out, int out_size, void* d_ws, size_t ws_size,
                              hipStream_t stream) {
    (void)in_sizes; (void)n_in; (void)out_size; (void)ws_size;
    const float* hX  = (const float*)d_in[0];   // [S, HID]
    const float* hWq = (const float*)d_in[1];   // [HID, NQH*HD]
    const float* hWk = (const float*)d_in[2];   // [HID, NKVH*HD]
    const float* hWv = (const float*)d_in[3];   // [HID, NKVH*HD]
    const float* hWo = (const float*)d_in[4];   // [NQH*HD, HID]
    float* out = (float*)d_out;                 // [S, HID]

    char* ws = (char*)d_ws;
    size_t off = 0;
    auto alloc = [&](size_t elems) { bf16_t* p = (bf16_t*)(ws + off);
                                     off += ((elems * sizeof(bf16_t) + 255) & ~(size_t)255);
                                     return p; };
    bf16_t* Xb  = alloc((size_t)S_LEN * HID);            // 16 MB
    bf16_t* WqT = alloc((size_t)HID * HID);              // 32 MB  [N=4096, K=4096]
    bf16_t* WkT = alloc((size_t)(NKVH*HD) * HID);        //  8 MB  [N=1024, K=4096]
    bf16_t* WvT = alloc((size_t)(NKVH*HD) * HID);        //  8 MB
    bf16_t* WoT = alloc((size_t)HID * HID);              // 32 MB  [N=4096, K=4096]
    bf16_t* Qb  = alloc((size_t)S_LEN * (NQH*HD));       // 16 MB
    bf16_t* Kb  = alloc((size_t)S_LEN * (NKVH*HD));      //  4 MB
    bf16_t* VTb = alloc((size_t)NKVH * HD * S_LEN);      //  4 MB
    bf16_t* Ab  = alloc((size_t)S_LEN * (NQH*HD));       // 16 MB   (~136 MB total)

    // 1) convert hidden states
    k_cvt_f32_bf16<<<2048, 256, 0, stream>>>(hX, Xb, S_LEN * HID);

    // 2) transpose+convert weights to [N, K] bf16
    dim3 tb(32, 8);
    k_transpose_f32_bf16<<<dim3(HID/32,       HID/32), tb, 0, stream>>>(hWq, WqT, HID, HID);
    k_transpose_f32_bf16<<<dim3((NKVH*HD)/32, HID/32), tb, 0, stream>>>(hWk, WkT, HID, NKVH*HD);
    k_transpose_f32_bf16<<<dim3((NKVH*HD)/32, HID/32), tb, 0, stream>>>(hWv, WvT, HID, NKVH*HD);
    k_transpose_f32_bf16<<<dim3(HID/32,       HID/32), tb, 0, stream>>>(hWo, WoT, HID, HID);

    // 3) projections (WMMA GEMMs): block tile 128x256, wave tile 64x64
    k_gemm_bf16<1><<<dim3((NQH*HD)/256,  S_LEN/128), 256, 0, stream>>>(Xb, WqT, Qb,
                    S_LEN, NQH*HD,  HID, NQH*HD);
    k_gemm_bf16<1><<<dim3((NKVH*HD)/256, S_LEN/128), 256, 0, stream>>>(Xb, WkT, Kb,
                    S_LEN, NKVH*HD, HID, NKVH*HD);
    k_gemm_bf16<2><<<dim3((NKVH*HD)/256, S_LEN/128), 256, 0, stream>>>(Xb, WvT, VTb,
                    S_LEN, NKVH*HD, HID, 0);

    // 4) flash attention: 32 heads * 128 q-tiles = 4096 waves / 8 per block
    k_flash_attn<<<(NQH * (S_LEN/16)) / 8, 256, 0, stream>>>(Qb, Kb, VTb, Ab);

    // 5) output projection -> f32 result
    k_gemm_bf16<0><<<dim3(HID/256, S_LEN/128), 256, 0, stream>>>(Ab, WoT, out,
                    S_LEN, HID, HID, HID);
}
